// RelationGCN_51110110822771
// MI455X (gfx1250) — matrix operand order repaired and checked by reading, not compiled, not run on hardware
//
#include <hip/hip_runtime.h>
#include <hip/hip_bf16.h>

// ---------------------------------------------------------------------------
// RelationGCN fused bf16-WMMA implementation for gfx1250 (MI455X, wave32).
// B=16, C=128, T=128, H=128.
// Round 2: hot relation-branch kernel now double-buffers its A-tiles through
// LDS with GLOBAL_LOAD_ASYNC_TO_LDS_B128 (ASYNCcnt pipeline), overlapping the
// 270 MB of HBM traffic with the bf16 WMMA stream.
// ---------------------------------------------------------------------------

#define Bq 16
#define Cq 128
#define Tq 128
#define Hq 128

typedef __attribute__((ext_vector_type(16))) __bf16 v16bf;
typedef __attribute__((ext_vector_type(8)))  __bf16 v8bf;
typedef __attribute__((ext_vector_type(8)))  float  v8f;
typedef __attribute__((ext_vector_type(4)))  float  v4f;

__device__ __forceinline__ v8f zero8() {
    v8f z;
#pragma unroll
    for (int i = 0; i < 8; ++i) z[i] = 0.0f;
    return z;
}

// A-fragment (16x32 bf16) from an LDS row of bf16.
// lane row = lane&15; half 0 -> K {k0..k0+15},
// half 1 -> K {k0+16..k0+31}.
__device__ __forceinline__ v16bf load_a_frag(const __bf16* rowp, int k0, int half) {
    v8bf lo = *(const v8bf*)(rowp + k0 + 16 * half);
    v8bf hi = *(const v8bf*)(rowp + k0 + 16 * half + 8);
    v16bf a;
#pragma unroll
    for (int i = 0; i < 8; ++i) { a[i] = lo[i]; a[8 + i] = hi[i]; }
    return a;
}

// A-fragment from an LDS row of f32 (async-staged tile): vector LDS loads
// then pack-convert to bf16 in VALU (co-executes with the WMMA pipe).
__device__ __forceinline__ v16bf load_a_frag_f32(const float* rowp, int k0, int half) {
    const float* p0 = rowp + k0 + 16 * half;
    const float* p1 = rowp + k0 + 16 * half + 8;
    v4f q0 = *(const v4f*)(p0);
    v4f q1 = *(const v4f*)(p0 + 4);
    v4f q2 = *(const v4f*)(p1);
    v4f q3 = *(const v4f*)(p1 + 4);
    v16bf a;
#pragma unroll
    for (int i = 0; i < 4; ++i) {
        a[i]      = (__bf16)q0[i];
        a[4 + i]  = (__bf16)q1[i];
        a[8 + i]  = (__bf16)q2[i];
        a[12 + i] = (__bf16)q3[i];
    }
    return a;
}

// ---------------------------------------------------------------------------
// Kernel 0: transpose-convert f32 weight (K x 128) into bf16 (128 x K).
// ---------------------------------------------------------------------------
__global__ __launch_bounds__(256) void convT_kernel(const float* __restrict__ W,
                                                    __bf16* __restrict__ Wt, int K) {
    int idx = blockIdx.x * 256 + threadIdx.x;   // over K*128
    int k = idx >> 7;
    int n = idx & 127;
    if (k < K) Wt[(size_t)n * K + k] = (__bf16)W[idx];
}

// ---------------------------------------------------------------------------
// Kernel 1: Out[2048,128] = A[2048,128] @ W (bf16, pre-transposed [N][K], K=128).
// ---------------------------------------------------------------------------
__global__ __launch_bounds__(256) void gemm_uv_kernel(const float* __restrict__ A,
                                                      const __bf16* __restrict__ Wt,
                                                      float* __restrict__ Out) {
    __shared__ __bf16 As[16][136];              // row stride 272B (16B multiple)
    const int tid = threadIdx.x;
    const int mbase = blockIdx.x * 16;
#pragma unroll
    for (int i = 0; i < 8; ++i) {
        int e = tid + i * 256;
        int r = e >> 7, k = e & 127;
        As[r][k] = (__bf16)A[(size_t)(mbase + r) * Hq + k];
    }
    __syncthreads();

    const int lane = tid & 31, wave = tid >> 5;
    const int half = lane >> 4, rrow = lane & 15;
    const int col = wave * 16 + rrow;

    v8f acc = zero8();
#pragma unroll
    for (int ks = 0; ks < 4; ++ks) {
        v16bf bfrag = *(const v16bf*)(Wt + (size_t)col * 128 + ks * 32 + 16 * half);
        v16bf afrag = load_a_frag(&As[rrow][0], ks * 32, half);
        acc = __builtin_amdgcn_wmma_f32_16x16x32_bf16(false, afrag, false, bfrag,
                                                      (short)0, acc, false, false);
    }
#pragma unroll
    for (int r = 0; r < 8; ++r) {
        int row = mbase + r + 8 * half;         // C/D layout: M = r + 8*half
        Out[(size_t)row * Hq + col] = acc[r];
    }
}

// ---------------------------------------------------------------------------
// Kernel 2: fused relation branch with async double-buffered A staging.
//   out[b,c,:] = max_t relu( rel[b,c,t,:]@W + U[b,c,:] (+ V[b,t,:]) + bias ) * mask[b,t]
// ---------------------------------------------------------------------------
#define ROWF 132                                // padded f32 row (528B, 16B multiple)

// Issue the async DMA for one 16x128 f32 tile into LDS buffer at ldsDst.
// 512 x 16B chunks, 2 per thread.
__device__ __forceinline__ void async_copy_tile(const float* __restrict__ gsrc,
                                                unsigned ldsDst, int tid) {
#pragma unroll
    for (int i = 0; i < 2; ++i) {
        int chunk = tid + i * 256;              // 0..511
        int r = chunk >> 5;                     // 32 chunks per row
        int kb = (chunk & 31) << 4;             // byte offset within row
        const float* g = gsrc + (size_t)r * Hq + (kb >> 2);
        unsigned l = ldsDst + (unsigned)(r * (ROWF * 4) + kb);
        asm volatile("global_load_async_to_lds_b128 %0, %1, off"
                     :: "v"(l), "v"(g) : "memory");
    }
}

__global__ __launch_bounds__(256) void branch_kernel(const float* __restrict__ Rel,
                                                     const __bf16* __restrict__ Wt,
                                                     const float* __restrict__ U,
                                                     const float* __restrict__ V,
                                                     const int*   __restrict__ Mask,
                                                     const float* __restrict__ bias,
                                                     float* __restrict__ Out) {
    __shared__ float Af[2][16][ROWF];           // 2 x 8.25KB double buffer
    __shared__ float mk[128];
    const int tid = threadIdx.x;
    const int bc = blockIdx.x;                  // b*C + c
    const int b = bc >> 7;
    const int lane = tid & 31, wave = tid >> 5;
    const int half = lane >> 4, rrow = lane & 15;
    const int col = wave * 16 + rrow;

    if (tid < 128) mk[tid] = (float)Mask[b * 128 + tid];
    const float ucol = U[(size_t)bc * Hq + col] + bias[col];

    // B fragments (16 columns x K=128) in registers for the whole block.
    v16bf bfrag[4];
#pragma unroll
    for (int ks = 0; ks < 4; ++ks)
        bfrag[ks] = *(const v16bf*)(Wt + (size_t)col * 128 + ks * 32 + 16 * half);

    const float* relBase = Rel + (size_t)bc * 128 * Hq;
    unsigned ldsBuf[2];
    ldsBuf[0] = (unsigned)(size_t)&Af[0][0][0]; // low 32 bits of generic ptr = LDS offset
    ldsBuf[1] = (unsigned)(size_t)&Af[1][0][0];

    v8f mx = zero8();                           // relu*mask >= 0 -> 0 is a valid floor

    // Prologue: kick off tile 0 DMA.
    async_copy_tile(relBase, ldsBuf[0], tid);

    for (int mt = 0; mt < 8; ++mt) {
        const int buf = mt & 1;
        // Prefetch next tile into the other buffer (its consumers synced last iter).
        if (mt + 1 < 8) {
            async_copy_tile(relBase + (size_t)(mt + 1) * 16 * Hq, ldsBuf[buf ^ 1], tid);
            asm volatile("s_wait_asynccnt 0x2" ::: "memory"); // current buf's 2 ops done
        } else {
            asm volatile("s_wait_asynccnt 0x0" ::: "memory");
        }
        __syncthreads();                        // every wave's chunks have landed

        v8f acc = zero8();
#pragma unroll
        for (int ks = 0; ks < 4; ++ks) {
            v16bf afrag = load_a_frag_f32(&Af[buf][rrow][0], ks * 32, half);
            acc = __builtin_amdgcn_wmma_f32_16x16x32_bf16(false, afrag, false, bfrag[ks],
                                                          (short)0, acc, false, false);
        }
#pragma unroll
        for (int r = 0; r < 8; ++r) {
            int t = mt * 16 + r + 8 * half;
            float val = acc[r] + ucol;
            if (V) val += V[((size_t)b * 128 + t) * Hq + col];
            val = fmaxf(val, 0.0f) * mk[t];
            mx[r] = fmaxf(mx[r], val);
        }
        __syncthreads();                        // compute done before buf is re-filled
    }

    float m = mx[0];
#pragma unroll
    for (int r = 1; r < 8; ++r) m = fmaxf(m, mx[r]);
    float other = __shfl_xor(m, 16, 32);        // combine row halves (lanes l <-> l^16)
    m = fmaxf(m, other);
    if (half == 0) Out[(size_t)bc * Hq + col] = m;
}

// ---------------------------------------------------------------------------
// Kernel 3: fused aggregation. Virtual concat [clone|food|thorn_agg|clone_agg]
// gathered into the LDS A tile (16 x 512 bf16); K=512 in 16 WMMA steps.
//   out = clone + relu(agg @ W_agg + b_agg)
// ---------------------------------------------------------------------------
__global__ __launch_bounds__(256) void agg_kernel(const float* __restrict__ clone,
                                                  const float* __restrict__ food,
                                                  const float* __restrict__ tagg,
                                                  const float* __restrict__ cagg,
                                                  const __bf16* __restrict__ WaT,
                                                  const float* __restrict__ bias,
                                                  float* __restrict__ Out) {
    __shared__ __bf16 As[16][520];              // row stride 1040B (16B multiple)
    const int tid = threadIdx.x;
    const int mbase = blockIdx.x * 16;
#pragma unroll
    for (int i = 0; i < 32; ++i) {
        int e = tid + i * 256;                  // 16*512 elements
        int r = e >> 9, k = e & 511;
        int sel = k >> 7, kk = k & 127;
        const float* s = (sel == 0) ? clone : (sel == 1) ? food : (sel == 2) ? tagg : cagg;
        As[r][k] = (__bf16)s[(size_t)(mbase + r) * Hq + kk];
    }
    __syncthreads();

    const int lane = tid & 31, wave = tid >> 5;
    const int half = lane >> 4, rrow = lane & 15;
    const int col = wave * 16 + rrow;

    v8f acc = zero8();
#pragma unroll
    for (int ks = 0; ks < 16; ++ks) {
        v16bf bfrag = *(const v16bf*)(WaT + (size_t)col * 512 + ks * 32 + 16 * half);
        v16bf afrag = load_a_frag(&As[rrow][0], ks * 32, half);
        acc = __builtin_amdgcn_wmma_f32_16x16x32_bf16(false, afrag, false, bfrag,
                                                      (short)0, acc, false, false);
    }
    const float bcol = bias[col];
#pragma unroll
    for (int r = 0; r < 8; ++r) {
        int row = mbase + r + 8 * half;
        Out[(size_t)row * Hq + col] =
            clone[(size_t)row * Hq + col] + fmaxf(acc[r] + bcol, 0.0f);
    }
}

// ---------------------------------------------------------------------------
extern "C" void kernel_launch(void* const* d_in, const int* in_sizes, int n_in,
                              void* d_out, int out_size, void* d_ws, size_t ws_size,
                              hipStream_t stream) {
    (void)in_sizes; (void)n_in; (void)out_size; (void)ws_size;

    const float* food       = (const float*)d_in[0];
    const float* thorn_rel  = (const float*)d_in[1];
    const float* clone      = (const float*)d_in[2];
    const float* clone_rel  = (const float*)d_in[3];
    const int*   thorn_mask = (const int*)d_in[4];
    const int*   clone_mask = (const int*)d_in[5];
    const float* W_thorn    = (const float*)d_in[6];
    const float* b_thorn    = (const float*)d_in[7];
    const float* W_clone    = (const float*)d_in[8];
    const float* b_clone    = (const float*)d_in[9];
    const float* W_agg      = (const float*)d_in[10];
    const float* b_agg      = (const float*)d_in[11];
    float* out = (float*)d_out;

    // Workspace carve-up (256B aligned slices).
    char* ws = (char*)d_ws;
    size_t off = 0;
    auto alloc = [&](size_t bytes) -> void* {
        off = (off + 255) & ~(size_t)255;
        void* p = ws + off;
        off += bytes;
        return p;
    };
    __bf16* WtT1 = (__bf16*)alloc(128 * 128 * sizeof(__bf16)); // W_thorn[:h]^T
    __bf16* WtT2 = (__bf16*)alloc(128 * 128 * sizeof(__bf16)); // W_thorn[h:]^T
    __bf16* WcT1 = (__bf16*)alloc(128 * 128 * sizeof(__bf16)); // W_clone[:h]^T
    __bf16* WcT2 = (__bf16*)alloc(128 * 128 * sizeof(__bf16)); // W_clone[h:2h]^T
    __bf16* WcT3 = (__bf16*)alloc(128 * 128 * sizeof(__bf16)); // W_clone[2h:]^T
    __bf16* WaT  = (__bf16*)alloc(128 * 512 * sizeof(__bf16)); // W_agg^T
    float* u_thorn   = (float*)alloc((size_t)Bq * Cq * Hq * sizeof(float));
    float* u_clone   = (float*)alloc((size_t)Bq * Cq * Hq * sizeof(float));
    float* v_clone   = (float*)alloc((size_t)Bq * Cq * Hq * sizeof(float));
    float* thorn_agg = (float*)alloc((size_t)Bq * Cq * Hq * sizeof(float));
    float* clone_agg = (float*)alloc((size_t)Bq * Cq * Hq * sizeof(float));

    // 0) weight conversion / transpose to bf16 [N][K]
    convT_kernel<<<64,  256, 0, stream>>>(W_thorn,            WtT1, 128);
    convT_kernel<<<64,  256, 0, stream>>>(W_thorn + 128*128,  WtT2, 128);
    convT_kernel<<<64,  256, 0, stream>>>(W_clone,            WcT1, 128);
    convT_kernel<<<64,  256, 0, stream>>>(W_clone + 128*128,  WcT2, 128);
    convT_kernel<<<64,  256, 0, stream>>>(W_clone + 256*128,  WcT3, 128);
    convT_kernel<<<256, 256, 0, stream>>>(W_agg,              WaT,  512);

    // 1) broadcast terms: clone @ W (2048 x 128 x 128)
    gemm_uv_kernel<<<128, 256, 0, stream>>>(clone, WtT1, u_thorn);
    gemm_uv_kernel<<<128, 256, 0, stream>>>(clone, WcT1, u_clone);
    gemm_uv_kernel<<<128, 256, 0, stream>>>(clone, WcT2, v_clone);

    // 2) fused relation branches (the two 8.6 GFLOP einsums + relu/mask/max),
    //    async double-buffered through LDS.
    branch_kernel<<<Bq * Cq, 256, 0, stream>>>(thorn_rel, WtT2, u_thorn, nullptr,
                                               thorn_mask, b_thorn, thorn_agg);
    branch_kernel<<<Bq * Cq, 256, 0, stream>>>(clone_rel, WcT3, u_clone, v_clone,
                                               clone_mask, b_clone, clone_agg);

    // 3) fused aggregation + residual
    agg_kernel<<<128, 256, 0, stream>>>(clone, food, thorn_agg, clone_agg,
                                        WaT, b_agg, out);
}
